// CrossScaleSelectiveScan_46926812676702
// MI455X (gfx1250) — matrix operand order repaired and checked by reading, not compiled, not run on hardware
//
#include <hip/hip_runtime.h>
#include <math.h>

#define DIMC 128
#define B_   8
#define H_   64
#define W_   64
#define NPIX (B_ * H_ * W_)   // 32768

typedef __bf16 bf16;
typedef bf16  bf16x16 __attribute__((ext_vector_type(16)));
typedef float floatx8 __attribute__((ext_vector_type(8)));

// ---------------------------------------------------------------------------
// CDNA5 async global->LDS path (guarded; falls back to synchronous staging)
// ---------------------------------------------------------------------------
#if defined(__gfx1250__) && __has_builtin(__builtin_amdgcn_global_load_async_to_lds_b128) && \
    __has_builtin(__builtin_amdgcn_s_wait_asynccnt)
#define USE_ASYNC_LDS 1
#else
#define USE_ASYNC_LDS 0
#endif

#if USE_ASYNC_LDS
// builtin expects pointers to gcc-style vector-of-4-int (per hipcc diagnostic)
typedef int v4i_async __attribute__((vector_size(4 * sizeof(int))));
__device__ __forceinline__ void async_b128(const void* g, void* l) {
  // generic->AS1/AS3 via integer casts (low 32 bits of a generic LDS pointer
  // are the LDS offset); emits global_load_async_to_lds_b128 (ASYNCcnt).
  __builtin_amdgcn_global_load_async_to_lds_b128(
      (__attribute__((address_space(1))) v4i_async*)(unsigned long long)g,
      (__attribute__((address_space(3))) v4i_async*)(unsigned int)(unsigned long long)l,
      0, 0);
}
#endif

// ---------------------------------------------------------------------------
// WMMA helpers (CDNA5 gfx1250, wave32)
// D = A(16x32 bf16) * B(32x16 bf16) + C(16x16 f32)
// ---------------------------------------------------------------------------
__device__ __forceinline__ floatx8 wmma_bf16(bf16x16 a, bf16x16 b, floatx8 c) {
  return __builtin_amdgcn_wmma_f32_16x16x32_bf16(
      /*neg_a=*/false, a, /*neg_b=*/false, b,
      /*c_mod=*/(short)0, c, /*reuse_a=*/false, /*reuse_b=*/false);
}

// A-matrix fragment: 16(M) x 32(K) bf16 tile from a row-major [rows][ld] array.
// ISA layout: lane L (L<16): M=L, K = {0..7} U {16..23}; lanes 16..31: K += 8.
__device__ __forceinline__ bf16x16 load_a_frag(const bf16* base, int ld, int k0, int lane) {
  bf16x16 a;
  int m   = lane & 15;
  int khi = (lane & 16) ? 8 : 0;
  const bf16* row = base + m * ld + k0 + khi;
#pragma unroll
  for (int e = 0; e < 16; ++e) {
    int k = (e < 8) ? e : (e + 8);
    a[e] = row[k];
  }
  return a;
}

// B-matrix fragment for out[n,o] = sum_k A[n,k] * W[o,k]  (B[k,o] = W[o,k]).
// ISA layout: lane L holds column N = L%16; K = {0..7} U {16..23} (lanes 0-15); lanes 16..31: K += 8.
__device__ __forceinline__ bf16x16 load_b_frag(const bf16* W, int ldw, int o0, int k0, int lane) {
  bf16x16 b;
  int o = o0 + (lane & 15);
  const bf16* row = W + (long)o * ldw + k0 + ((lane & 16) ? 8 : 0);
#pragma unroll
  for (int e = 0; e < 16; ++e) b[e] = row[(e < 8) ? e : (e + 8)];
  return b;
}

__device__ __forceinline__ float sigmoidf_(float x) {
  return 1.0f / (1.0f + __expf(-x));
}

// ---------------------------------------------------------------------------
// K0: f32 -> bf16 weight conversion
// ---------------------------------------------------------------------------
__global__ void cvt_f32_bf16(const float* __restrict__ src, bf16* __restrict__ dst, int n) {
  int i = blockIdx.x * blockDim.x + threadIdx.x;
  if (i < n) dst[i] = (bf16)src[i];
}

// ---------------------------------------------------------------------------
// K1: bilinear resize (l down 2x antialiased, s up 2x) + concat -> xin bf16
//     xin layout: [B, H, W, 3C] channel-last (GEMM A operand)
// ---------------------------------------------------------------------------
__global__ void build_xin_kernel(const float* __restrict__ l, const float* __restrict__ m,
                                 const float* __restrict__ s, bf16* __restrict__ xin) {
  int x = threadIdx.x;   // 0..63
  int y = blockIdx.x;    // 0..63
  int c = blockIdx.y;    // 0..127
  int b = blockIdx.z;    // 0..7

  // l: [B,128,128,128], antialiased bilinear downsample: taps [1,3,3,1]/8 per dim
  const float wt[4] = {0.125f, 0.375f, 0.375f, 0.125f};
  const float* lp = l + (long)(b * DIMC + c) * 128 * 128;
  float lr = 0.0f;
#pragma unroll
  for (int i = 0; i < 4; ++i) {
    int sy = min(max(2 * y - 1 + i, 0), 127);
#pragma unroll
    for (int j = 0; j < 4; ++j) {
      int sx = min(max(2 * x - 1 + j, 0), 127);
      lr += wt[i] * wt[j] * lp[sy * 128 + sx];
    }
  }

  float mm = m[((long)(b * DIMC + c) * 64 + y) * 64 + x];

  // s: [B,128,32,32], bilinear upsample 2x (align_corners=False)
  const float* sp = s + (long)(b * DIMC + c) * 32 * 32;
  int iy = y >> 1, ix = x >> 1;
  int y0, y1, x0, x1;
  float wy0, wy1, wx0, wx1;
  if (y & 1) { y0 = iy; y1 = min(iy + 1, 31); wy0 = 0.75f; wy1 = 0.25f; }
  else       { y0 = max(iy - 1, 0); y1 = iy;  wy0 = 0.25f; wy1 = 0.75f; }
  if (x & 1) { x0 = ix; x1 = min(ix + 1, 31); wx0 = 0.75f; wx1 = 0.25f; }
  else       { x0 = max(ix - 1, 0); x1 = ix;  wx0 = 0.25f; wx1 = 0.75f; }
  float sr = wy0 * (wx0 * sp[y0 * 32 + x0] + wx1 * sp[y0 * 32 + x1]) +
             wy1 * (wx0 * sp[y1 * 32 + x0] + wx1 * sp[y1 * 32 + x1]);

  long p = ((long)(b * H_ + y) * W_ + x) * 384;
  xin[p + c]        = (bf16)lr;
  xin[p + 128 + c]  = (bf16)mm;
  xin[p + 256 + c]  = (bf16)sr;
}

// ---------------------------------------------------------------------------
// K2: proj_in GEMM [32768,384]x[384->128] + BN + ReLU -> x (f32 and bf16)
//     block = 16 rows x 128 cols; 8 waves, one 16x16 col-tile each; K=12x32
// ---------------------------------------------------------------------------
__global__ __launch_bounds__(256) void proj_in_kernel(
    const bf16* __restrict__ xin, const bf16* __restrict__ wpin,
    const float* __restrict__ scale, const float* __restrict__ shift,
    float* __restrict__ xf, bf16* __restrict__ xbf) {
  __shared__ alignas(16) bf16 sA[16][384];   // 12 KB, contiguous with global tile
  int tid = threadIdx.x, lane = tid & 31, wv = tid >> 5;
  long n0 = (long)blockIdx.x * 16;

#if USE_ASYNC_LDS
  // tile is 16 rows x 384 bf16 = 12288 B, contiguous in global: 3 x b128/thread
#pragma unroll
  for (int j = 0; j < 3; ++j) {
    int eoff = tid * 8 + j * 2048;           // bf16 elements (16 B chunks)
    async_b128(xin + n0 * 384 + eoff, &sA[0][0] + eoff);
  }
  __builtin_amdgcn_s_wait_asynccnt(0);
#else
  for (int i = tid; i < 16 * 384; i += 256) {
    int mm = i / 384, k = i % 384;
    sA[mm][k] = xin[(n0 + mm) * 384 + k];
  }
#endif
  __syncthreads();

  floatx8 acc = {};
  int o0 = wv * 16;
#pragma unroll
  for (int kb = 0; kb < 12; ++kb) {
    bf16x16 a  = load_a_frag(&sA[0][0], 384, kb * 32, lane);
    bf16x16 bw = load_b_frag(wpin, 384, o0, kb * 32, lane);
    acc = wmma_bf16(a, bw, acc);
  }

  int hi8 = (lane & 16) ? 8 : 0;
  int o = o0 + (lane & 15);
  float sc_ = scale[o], sh_ = shift[o];
#pragma unroll
  for (int v = 0; v < 8; ++v) {
    float y = fmaxf(acc[v] * sc_ + sh_, 0.0f);
    long idx = (n0 + v + hi8) * DIMC + o;
    xf[idx]  = y;
    xbf[idx] = (bf16)y;
  }
}

// ---------------------------------------------------------------------------
// K3/K4: GRU scan (persistent WG, register-resident bf16 weight fragments,
//   double-buffered async x_t staging).
//   VERT: sequence over H, rows = (b, w0..w0+15). else over W, rows = (b, h..).
//   Each step: gx = x_t*Wih^T, gh = h*Whh^T (two [16,128]x[128,384] WMMA GEMMs),
//   then r|z|n combine; h written to `scn` (= for VERT, += for horizontal).
// ---------------------------------------------------------------------------
template <bool VERT, bool ACCUM>
__global__ __launch_bounds__(256) void scan_kernel(
    const bf16* __restrict__ xb, const bf16* __restrict__ wih,
    const bf16* __restrict__ whh, const float* __restrict__ bih,
    const float* __restrict__ bhh, float* __restrict__ scn) {
  __shared__ alignas(16) bf16  sA[2][16][128]; // double-buffered x_t tile
  __shared__ alignas(16) bf16  sH[16][128];    // hidden state (bf16 A operand)
  __shared__ float sGX[16][384];               // gx accumulators
  __shared__ float sGH[16][384];               // gh accumulators
  __shared__ float sBih[384];
  __shared__ float sBhh[384];

  int tid = threadIdx.x, lane = tid & 31, wv = tid >> 5;
  int seg = blockIdx.x;             // 0..31, 16 batch rows each
  int b   = seg >> 2;
  int off = (seg & 3) << 4;         // w0 (VERT) or h0 (horizontal)

  long base; int strideM, strideT;  // elem index = base + m*strideM + t*strideT + c
  if (VERT) { base = (long)b * H_ * W_ * DIMC + (long)off * DIMC;       strideM = DIMC;       strideT = W_ * DIMC; }
  else      { base = (long)b * H_ * W_ * DIMC + (long)off * W_ * DIMC;  strideM = W_ * DIMC;  strideT = DIMC; }

  // per-thread async-copy coordinates: one 16B chunk of the 16x128 bf16 tile
  int cm = tid >> 4;                // row 0..15
  int ck = (tid & 15) * 8;          // k offset, 16B aligned

  // Register-resident weight fragments: per wave 3 col-tiles x 4 K-blocks x 2 mats.
  bf16x16 Bih[3][4], Bhh[3][4];
#pragma unroll
  for (int ct = 0; ct < 3; ++ct) {
    int o0 = (ct * 8 + wv) * 16;
#pragma unroll
    for (int kb = 0; kb < 4; ++kb) {
      Bih[ct][kb] = load_b_frag(wih, DIMC, o0, kb * 32, lane);
      Bhh[ct][kb] = load_b_frag(whh, DIMC, o0, kb * 32, lane);
    }
  }

  for (int i = tid; i < 384; i += 256) { sBih[i] = bih[i]; sBhh[i] = bhh[i]; }
  for (int i = tid; i < 16 * DIMC; i += 256) sH[i >> 7][i & 127] = (bf16)0.0f;

#if USE_ASYNC_LDS
  // prefetch t=0 into buffer 0
  async_b128(xb + base + (long)cm * strideM + ck, &sA[0][cm][ck]);
#endif
  __syncthreads();

  int cur = 0;
  for (int t = 0; t < H_; ++t) {
#if USE_ASYNC_LDS
    if (t + 1 < H_) {
      // overlap: issue t+1 into the other buffer, wait for current tile only
      async_b128(xb + base + (long)cm * strideM + (long)(t + 1) * strideT + ck,
                 &sA[cur ^ 1][cm][ck]);
      __builtin_amdgcn_s_wait_asynccnt(1);
    } else {
      __builtin_amdgcn_s_wait_asynccnt(0);
    }
#else
    for (int i = tid; i < 16 * DIMC; i += 256) {
      int mm = i >> 7, k = i & 127;
      sA[cur][mm][k] = xb[base + (long)mm * strideM + (long)t * strideT + k];
    }
#endif
    __syncthreads();

    floatx8 accX[3] = {}, accH[3] = {};
#pragma unroll
    for (int kb = 0; kb < 4; ++kb) {
      bf16x16 aX = load_a_frag(&sA[cur][0][0], 128, kb * 32, lane);
      bf16x16 aH = load_a_frag(&sH[0][0], 128, kb * 32, lane);
#pragma unroll
      for (int ct = 0; ct < 3; ++ct) {
        accX[ct] = wmma_bf16(aX, Bih[ct][kb], accX[ct]);
        accH[ct] = wmma_bf16(aH, Bhh[ct][kb], accH[ct]);
      }
    }

    int hi8 = (lane & 16) ? 8 : 0;
    int oc  = lane & 15;
#pragma unroll
    for (int ct = 0; ct < 3; ++ct) {
      int o0 = (ct * 8 + wv) * 16;
#pragma unroll
      for (int v = 0; v < 8; ++v) {
        sGX[v + hi8][o0 + oc] = accX[ct][v];
        sGH[v + hi8][o0 + oc] = accH[ct][v];
      }
    }
    __syncthreads();

    // elementwise GRU combine (r | z | n layout), write h to LDS + global
    for (int i = tid; i < 16 * DIMC; i += 256) {
      int mm = i >> 7, c = i & 127;
      float r  = sigmoidf_(sGX[mm][c]       + sBih[c]       + sGH[mm][c]       + sBhh[c]);
      float z  = sigmoidf_(sGX[mm][c + 128] + sBih[c + 128] + sGH[mm][c + 128] + sBhh[c + 128]);
      float nn = tanhf(sGX[mm][c + 256] + sBih[c + 256] + r * (sGH[mm][c + 256] + sBhh[c + 256]));
      float hprev = (float)sH[mm][c];
      float hnew  = (1.0f - z) * nn + z * hprev;
      sH[mm][c] = (bf16)hnew;
      long gi = base + (long)mm * strideM + (long)t * strideT + c;
      if (ACCUM) scn[gi] += hnew; else scn[gi] = hnew;
    }
    __syncthreads();
    cur ^= 1;
  }
}

// ---------------------------------------------------------------------------
// K5: gate GEMM + sigmoid-gating + proj_out GEMM + BN + ReLU + residual
//     two chained WMMA GEMMs per 16-row tile through LDS; writes [B,C,H,W].
// ---------------------------------------------------------------------------
__global__ __launch_bounds__(256) void gate_out_kernel(
    const float* __restrict__ scn, const bf16* __restrict__ wgate,
    const float* __restrict__ gate_b, const bf16* __restrict__ wpout,
    const float* __restrict__ po_scale, const float* __restrict__ po_shift,
    const float* __restrict__ xf, float* __restrict__ out) {
  __shared__ alignas(16) float sS[16][128];    // scanned tile (contiguous in global)
  __shared__ alignas(16) bf16  sSb[16][128];
  __shared__ alignas(16) bf16  sG[16][128];
  int tid = threadIdx.x, lane = tid & 31, wv = tid >> 5;
  long n0 = (long)blockIdx.x * 16;

#if USE_ASYNC_LDS
  // tile = 16x128 f32 = 8192 B contiguous: 2 x b128 per thread
#pragma unroll
  for (int j = 0; j < 2; ++j) {
    int foff = tid * 4 + j * 1024;             // float elements (16 B chunks)
    async_b128(scn + n0 * DIMC + foff, &sS[0][0] + foff);
  }
  __builtin_amdgcn_s_wait_asynccnt(0);
  __syncthreads();
  for (int i = tid; i < 16 * 128; i += 256) {
    int mm = i >> 7, c = i & 127;
    sSb[mm][c] = (bf16)sS[mm][c];
  }
#else
  for (int i = tid; i < 16 * 128; i += 256) {
    int mm = i >> 7, c = i & 127;
    float v = scn[(n0 + mm) * DIMC + c];
    sS[mm][c]  = v;
    sSb[mm][c] = (bf16)v;
  }
#endif
  __syncthreads();

  int o0  = wv * 16;
  int hi8 = (lane & 16) ? 8 : 0;
  int o   = o0 + (lane & 15);

  floatx8 acc = {};
#pragma unroll
  for (int kb = 0; kb < 4; ++kb) {
    bf16x16 a  = load_a_frag(&sSb[0][0], 128, kb * 32, lane);
    bf16x16 bw = load_b_frag(wgate, 128, o0, kb * 32, lane);
    acc = wmma_bf16(a, bw, acc);
  }
  float gb = gate_b[o];
#pragma unroll
  for (int v = 0; v < 8; ++v) {
    int mm = v + hi8;
    float g = sigmoidf_(acc[v] + gb);
    sG[mm][o] = (bf16)(sS[mm][o] * g);   // each (mm,o) written by exactly one lane
  }
  __syncthreads();

  floatx8 acc2 = {};
#pragma unroll
  for (int kb = 0; kb < 4; ++kb) {
    bf16x16 a  = load_a_frag(&sG[0][0], 128, kb * 32, lane);
    bf16x16 bw = load_b_frag(wpout, 128, o0, kb * 32, lane);
    acc2 = wmma_bf16(a, bw, acc2);
  }
  float psc = po_scale[o], psh = po_shift[o];
#pragma unroll
  for (int v = 0; v < 8; ++v) {
    int  mm = v + hi8;
    long n  = n0 + mm;
    float y = fmaxf(acc2[v] * psc + psh, 0.0f) + xf[n * DIMC + o];
    int bb = (int)(n >> 12);
    int hh = (int)((n >> 6) & 63);
    int ww = (int)(n & 63);
    out[(((long)bb * DIMC + o) * H_ + hh) * W_ + ww] = y;  // [B,C,H,W]
  }
}

// ---------------------------------------------------------------------------
// Host-side orchestration
// ---------------------------------------------------------------------------
extern "C" void kernel_launch(void* const* d_in, const int* in_sizes, int n_in,
                              void* d_out, int out_size, void* d_ws, size_t ws_size,
                              hipStream_t stream) {
  (void)in_sizes; (void)n_in; (void)out_size; (void)ws_size;
  const float* l          = (const float*)d_in[0];
  const float* m          = (const float*)d_in[1];
  const float* s          = (const float*)d_in[2];
  const float* proj_in_w  = (const float*)d_in[3];
  const float* pin_scale  = (const float*)d_in[4];
  const float* pin_shift  = (const float*)d_in[5];
  const float* wih_h      = (const float*)d_in[6];
  const float* whh_h      = (const float*)d_in[7];
  const float* bih_h      = (const float*)d_in[8];
  const float* bhh_h      = (const float*)d_in[9];
  const float* wih_w      = (const float*)d_in[10];
  const float* whh_w      = (const float*)d_in[11];
  const float* bih_w      = (const float*)d_in[12];
  const float* bhh_w      = (const float*)d_in[13];
  const float* gate_w     = (const float*)d_in[14];
  const float* gate_b     = (const float*)d_in[15];
  const float* proj_out_w = (const float*)d_in[16];
  const float* po_scale   = (const float*)d_in[17];
  const float* po_shift   = (const float*)d_in[18];
  float* out = (float*)d_out;

  char* ws = (char*)d_ws;
  // bf16 weight pool (element offsets)
  bf16* wbf     = (bf16*)ws;
  bf16* pin_bf  = wbf + 0;        // 128x384
  bf16* wihh_bf = wbf + 49152;    // 384x128
  bf16* whhh_bf = wbf + 98304;    // 384x128
  bf16* wihw_bf = wbf + 147456;   // 384x128
  bf16* whhw_bf = wbf + 196608;   // 384x128
  bf16* gate_bf = wbf + 245760;   // 128x128
  bf16* pout_bf = wbf + 262144;   // 128x128  (ends at byte 557056)
  bf16*  xin = (bf16*) (ws + 557056);     // [32768,384] bf16  (24 MB)
  float* xf  = (float*)(ws + 25722880);   // [32768,128] f32   (16 MB)
  bf16*  xbf = (bf16*) (ws + 42500096);   // [32768,128] bf16  ( 8 MB)
  float* sc  = (float*)(ws + 50888704);   // [32768,128] f32   (16 MB)

  // K0: weight conversions
  cvt_f32_bf16<<<(49152 + 255) / 256, 256, 0, stream>>>(proj_in_w,  pin_bf,  49152);
  cvt_f32_bf16<<<(49152 + 255) / 256, 256, 0, stream>>>(wih_h,      wihh_bf, 49152);
  cvt_f32_bf16<<<(49152 + 255) / 256, 256, 0, stream>>>(whh_h,      whhh_bf, 49152);
  cvt_f32_bf16<<<(49152 + 255) / 256, 256, 0, stream>>>(wih_w,      wihw_bf, 49152);
  cvt_f32_bf16<<<(49152 + 255) / 256, 256, 0, stream>>>(whh_w,      whhw_bf, 49152);
  cvt_f32_bf16<<<(16384 + 255) / 256, 256, 0, stream>>>(gate_w,     gate_bf, 16384);
  cvt_f32_bf16<<<(16384 + 255) / 256, 256, 0, stream>>>(proj_out_w, pout_bf, 16384);

  // K1: resize + concat -> xin
  build_xin_kernel<<<dim3(H_, DIMC, B_), 64, 0, stream>>>(l, m, s, xin);

  // K2: proj_in + BN + ReLU -> x (f32 + bf16)
  proj_in_kernel<<<NPIX / 16, 256, 0, stream>>>(xin, pin_bf, pin_scale, pin_shift, xf, xbf);

  // K3: vertical GRU scan (writes scanned)
  scan_kernel<true, false><<<32, 256, 0, stream>>>(xbf, wihh_bf, whhh_bf, bih_h, bhh_h, sc);

  // K4: horizontal GRU scan (accumulates into scanned)
  scan_kernel<false, true><<<32, 256, 0, stream>>>(xbf, wihw_bf, whhw_bf, bih_w, bhh_w, sc);

  // K5: gate + proj_out + BN + ReLU + residual -> out [B,C,H,W]
  gate_out_kernel<<<NPIX / 16, 256, 0, stream>>>(sc, gate_bf, gate_b, pout_bf,
                                                 po_scale, po_shift, xf, out);
}
